// HEALUpSampler_46377056863018
// MI455X (gfx1250) — matrix operand order repaired and checked by reading, not compiled, not run on hardware
//
#include <hip/hip_runtime.h>
#include <hip/hip_bf16.h>

// ---------------- problem constants ----------------
#define N_SEND 12288
#define N_REC  49152
#define E_TOT  196608
#define D_DIM  256
#define B_BATCH 2

typedef __attribute__((ext_vector_type(16))) __bf16 v16bf;
typedef __attribute__((ext_vector_type(8)))  __bf16 v8bf;
typedef __attribute__((ext_vector_type(4)))  __bf16 v4bf;
typedef __attribute__((ext_vector_type(8)))  float  v8f;
typedef __attribute__((ext_vector_type(4)))  float  v4f;

__device__ __forceinline__ float gelu_tanh(float x) {
    // jax.nn.gelu default (approximate=True, tanh form)
    float x3 = x * x * x;
    float t  = tanhf(0.7978845608028654f * (x + 0.044715f * x3));
    return 0.5f * x * (1.0f + t);
}

// ---------------------------------------------------------------------------
// Kernel 0: convert + transpose weights to bf16, N-major (row n holds K entries
// contiguously) so B-fragments are two 16B loads per lane (k = h*8.. and 16+h*8..).
// We2T: 256x256, Wl1T: 256x512, Wl2T: 256x256
// ---------------------------------------------------------------------------
__global__ void prep_weights_kernel(const float* __restrict__ We2,
                                    const float* __restrict__ Wl1,
                                    const float* __restrict__ Wl2,
                                    __bf16* __restrict__ We2T,
                                    __bf16* __restrict__ Wl1T,
                                    __bf16* __restrict__ Wl2T) {
    int i = blockIdx.x * blockDim.x + threadIdx.x;
    if (i < 65536) {                       // We2T[n*256+k] = We2[k*256+n]
        int n = i >> 8, k = i & 255;
        We2T[i] = (__bf16)We2[k * 256 + n];
    } else if (i < 65536 + 131072) {       // Wl1T[n*512+k] = Wl1[k*256+n]
        int j = i - 65536;
        int n = j >> 9, k = j & 511;
        Wl1T[j] = (__bf16)Wl1[k * 256 + n];
    } else if (i < 262144) {               // Wl2T[n*256+k] = Wl2[k*256+n]
        int j = i - 196608;
        int n = j >> 8, k = j & 255;
        Wl2T[j] = (__bf16)Wl2[k * 256 + n];
    }
}

// ---------------------------------------------------------------------------
// Kernel 1: segment offsets via binary search (idx_rec is sorted).
// seg[r] = lower_bound(idx_rec, r); seg[N_REC] == E.
// ---------------------------------------------------------------------------
__global__ void seg_bounds_kernel(const int* __restrict__ idx_rec,
                                  int* __restrict__ seg) {
    int r = blockIdx.x * blockDim.x + threadIdx.x;
    if (r > N_REC) return;
    int lo = 0, hi = E_TOT;
    while (lo < hi) {
        int mid = (lo + hi) >> 1;
        if (idx_rec[mid] < r) lo = mid + 1; else hi = mid;
    }
    seg[r] = lo;
}

// ---------------------------------------------------------------------------
// Kernel 2: edge MLP.
// h1 = gelu(edge_attr @ We1 + be1) computed directly into WMMA A-fragment
// layout (16-bit A 16x32: lane half h holds K in {h*8..+8, 16+h*8..+8}),
// then ef = h1 @ We2 + be2 via v_wmma_f32_16x16x32_bf16, stored bf16.
// Block = 256 threads = 8 waves; each wave owns a 16-edge row tile.
// ---------------------------------------------------------------------------
__global__ void __launch_bounds__(256)
__attribute__((amdgpu_waves_per_eu(2)))
edge_mlp_kernel(const float* __restrict__ edge_attr,  // [E][4]
                const float* __restrict__ We1,        // [4][256]
                const float* __restrict__ be1,        // [256]
                const __bf16* __restrict__ We2T,      // [256][256] N-major
                const float* __restrict__ be2,        // [256]
                __bf16* __restrict__ ef) {            // [E][256]
    __shared__ float4 sW1[256];   // We1 column-major: per output col c, k=0..3
    __shared__ float  sB1[256];
    __shared__ float  sB2[256];

    int tid = threadIdx.x;
    sW1[tid] = make_float4(We1[0 * 256 + tid], We1[1 * 256 + tid],
                           We1[2 * 256 + tid], We1[3 * 256 + tid]);
    sB1[tid] = be1[tid];
    sB2[tid] = be2[tid];
    __syncthreads();

    int wave = tid >> 5;
    int lane = tid & 31;
    int m = lane & 15;      // A row / B-C column-within-tile
    int h = lane >> 4;      // lane half selects K sub-chunks

    long e0  = (long)blockIdx.x * 128 + wave * 16;
    long row = e0 + m;
    float4 ea = *(const float4*)(edge_attr + (size_t)row * 4);

    // Build the 8 A-fragments (K = 256 -> 8 k-tiles of 32) in registers.
    v16bf afrag[8];
#pragma unroll
    for (int kt = 0; kt < 8; ++kt) {
#pragma unroll
        for (int i = 0; i < 16; ++i) {
            int c = kt * 32 + h * 8 + i + (i >= 8 ? 8 : 0);
            float4 w = sW1[c];
            float v = ea.x * w.x + ea.y * w.y + ea.z * w.z + ea.w * w.w + sB1[c];
            afrag[kt][i] = (__bf16)gelu_tanh(v);
        }
    }

    // GEMM: 16 column tiles of We2. nt unrolled x2 -> two independent
    // accumulate chains so B loads of one hide behind WMMAs of the other.
#pragma unroll 2
    for (int nt = 0; nt < 16; ++nt) {
        v8f acc = {};
        int n = nt * 16 + m;
        const __bf16* bp = We2T + (size_t)n * 256 + h * 8;
#pragma unroll
        for (int kt = 0; kt < 8; ++kt) {
            v8bf blo = *(const v8bf*)(bp + kt * 32);
            v8bf bhi = *(const v8bf*)(bp + kt * 32 + 16);
            v16bf b = __builtin_shufflevector(blo, bhi, 0, 1, 2, 3, 4, 5, 6, 7, 8, 9, 10, 11, 12, 13, 14, 15);
            acc = __builtin_amdgcn_wmma_f32_16x16x32_bf16(
                false, afrag[kt], false, b, (short)0, acc, false, false);
        }
        float bias = sB2[n];
        // C/D layout: row = v + 8*h, col = n. Single base + immediate offsets.
        __bf16* op = ef + (size_t)(e0 + 8 * h) * 256 + n;
#pragma unroll
        for (int v = 0; v < 8; ++v) {
            op[v * 256] = (__bf16)(acc[v] + bias);
        }
    }
}

// ---------------------------------------------------------------------------
// Kernel 3: gather + concat + segment-sum, no atomics (idx_rec sorted ->
// contiguous edge range per receiver). One 64-thread block per receiver,
// 4 channels/thread (b128 loads); both batches fused so ef is read once.
// x (25MB) + ef (100MB bf16) stay L2-resident; agg is streamed with NT stores
// so it does not evict them.
// ---------------------------------------------------------------------------
__global__ void gather_segsum_kernel(const float* __restrict__ x,       // [B][N_SEND][256]
                                     const int* __restrict__ idx_send,  // [E]
                                     const __bf16* __restrict__ ef,     // [E][256]
                                     const int* __restrict__ seg,       // [N_REC+1]
                                     float* __restrict__ agg) {         // [B][N_REC][512]
    int r = blockIdx.x;
    int c = threadIdx.x * 4;              // 64 threads, 4 channels each
    int s0 = seg[r], s1 = seg[r + 1];

    v4f aX0 = {0.f, 0.f, 0.f, 0.f};
    v4f aX1 = {0.f, 0.f, 0.f, 0.f};
    v4f aE  = {0.f, 0.f, 0.f, 0.f};
    for (int e = s0; e < s1; ++e) {
        int s = idx_send[e];
        v4f x0 = *(const v4f*)(x + (size_t)s * 256 + c);
        v4f x1 = *(const v4f*)(x + (size_t)(N_SEND + s) * 256 + c);
        v4bf ev = *(const v4bf*)(ef + (size_t)e * 256 + c);
        aX0 += x0;
        aX1 += x1;
#pragma unroll
        for (int i = 0; i < 4; ++i) aE[i] += (float)ev[i];
    }
    size_t o0 = (size_t)r * 512;
    size_t o1 = (size_t)(N_REC + r) * 512;
    __builtin_nontemporal_store(aX0, (v4f*)(agg + o0 + c));
    __builtin_nontemporal_store(aE,  (v4f*)(agg + o0 + 256 + c));
    __builtin_nontemporal_store(aX1, (v4f*)(agg + o1 + c));
    __builtin_nontemporal_store(aE,  (v4f*)(agg + o1 + 256 + c));
}

// ---------------------------------------------------------------------------
// Kernel 4: node MLP. out = gelu(agg @ Wl1 + bl1) @ Wl2 + bl2.
// Block = 128 threads = 4 waves, 16 rows per wave (64 rows per block).
// A1 (K=512 -> 16 k-tiles) register-resident; intermediate re-laid
// C-layout -> A-layout through LDS for GEMM2. agg/out streamed non-temporal.
// ---------------------------------------------------------------------------
__global__ void __launch_bounds__(128)
__attribute__((amdgpu_waves_per_eu(2)))
node_mlp_kernel(const float* __restrict__ agg,    // [B*N_REC][512]
                const __bf16* __restrict__ Wl1T,  // [256][512] N-major
                const float* __restrict__ bl1,
                const __bf16* __restrict__ Wl2T,  // [256][256] N-major
                const float* __restrict__ bl2,
                float* __restrict__ out) {        // [B*N_REC][256]
    __shared__ __bf16 sH[4][16 * 256];    // per-wave 16x256 intermediate (32 KB)
    __shared__ float  sB1[256];
    __shared__ float  sB2[256];

    int tid = threadIdx.x;                // 128
    sB1[tid]       = bl1[tid];
    sB1[tid + 128] = bl1[tid + 128];
    sB2[tid]       = bl2[tid];
    sB2[tid + 128] = bl2[tid + 128];
    __syncthreads();

    int wave = tid >> 5;
    int lane = tid & 31;
    int m = lane & 15;
    int h = lane >> 4;

    size_t r0 = (size_t)blockIdx.x * 64 + (size_t)wave * 16;

    // Load A1 fragments from agg (f32 -> bf16), 16 k-tiles of 32, via b128.
    v16bf a1[16];
    const float* arow = agg + (r0 + m) * 512;
#pragma unroll
    for (int kt = 0; kt < 16; ++kt) {
        const float* p = arow + kt * 32 + h * 8;
        v4f q0 = __builtin_nontemporal_load((const v4f*)(p));
        v4f q1 = __builtin_nontemporal_load((const v4f*)(p + 4));
        v4f q2 = __builtin_nontemporal_load((const v4f*)(p + 16));
        v4f q3 = __builtin_nontemporal_load((const v4f*)(p + 20));
#pragma unroll
        for (int i = 0; i < 4; ++i) {
            a1[kt][i]      = (__bf16)q0[i];   // K = kt*32 + h*8 + i
            a1[kt][4 + i]  = (__bf16)q1[i];
            a1[kt][8 + i]  = (__bf16)q2[i];   // K = kt*32 + 16 + h*8 + i
            a1[kt][12 + i] = (__bf16)q3[i];
        }
    }

    __bf16* hrow = &sH[wave][0];

    // GEMM1 + bias + gelu -> LDS (bf16, row-major 16x256 per wave)
#pragma unroll 2
    for (int nt = 0; nt < 16; ++nt) {
        v8f acc = {};
        int n = nt * 16 + m;
        const __bf16* bp = Wl1T + (size_t)n * 512 + h * 8;
#pragma unroll
        for (int kt = 0; kt < 16; ++kt) {
            v8bf blo = *(const v8bf*)(bp + kt * 32);
            v8bf bhi = *(const v8bf*)(bp + kt * 32 + 16);
            v16bf b = __builtin_shufflevector(blo, bhi, 0, 1, 2, 3, 4, 5, 6, 7, 8, 9, 10, 11, 12, 13, 14, 15);
            acc = __builtin_amdgcn_wmma_f32_16x16x32_bf16(
                false, a1[kt], false, b, (short)0, acc, false, false);
        }
        float bias = sB1[n];
        __bf16* hp = hrow + (8 * h) * 256 + n;
#pragma unroll
        for (int v = 0; v < 8; ++v) {
            hp[v * 256] = (__bf16)gelu_tanh(acc[v] + bias);
        }
    }
    __syncthreads();

    // Re-read intermediate as A-fragments for GEMM2 (K2 = 256 -> 8 k-tiles).
    v16bf a2[8];
#pragma unroll
    for (int kt = 0; kt < 8; ++kt) {
        const __bf16* p = hrow + m * 256 + kt * 32 + h * 8;
        v8bf lo = *(const v8bf*)(p);
        v8bf hi = *(const v8bf*)(p + 16);
        a2[kt] = __builtin_shufflevector(lo, hi, 0, 1, 2, 3, 4, 5, 6, 7,
                                         8, 9, 10, 11, 12, 13, 14, 15);
    }

    // GEMM2 + bias -> out (f32, streamed NT)
#pragma unroll 2
    for (int nt = 0; nt < 16; ++nt) {
        v8f acc = {};
        int n = nt * 16 + m;
        const __bf16* bp = Wl2T + (size_t)n * 256 + h * 8;
#pragma unroll
        for (int kt = 0; kt < 8; ++kt) {
            v8bf blo = *(const v8bf*)(bp + kt * 32);
            v8bf bhi = *(const v8bf*)(bp + kt * 32 + 16);
            v16bf b = __builtin_shufflevector(blo, bhi, 0, 1, 2, 3, 4, 5, 6, 7, 8, 9, 10, 11, 12, 13, 14, 15);
            acc = __builtin_amdgcn_wmma_f32_16x16x32_bf16(
                false, a2[kt], false, b, (short)0, acc, false, false);
        }
        float bias = sB2[n];
        float* op = out + (r0 + 8 * h) * 256 + n;
#pragma unroll
        for (int v = 0; v < 8; ++v) {
            __builtin_nontemporal_store(acc[v] + bias, op + v * 256);
        }
    }
}

// ---------------------------------------------------------------------------
// Host-side launcher.
// Inputs (setup_inputs order): x, edge_attr, idx_send, idx_rec,
//   We1, be1, We2, be2, Wl1, bl1, Wl2, bl2
// Workspace layout (bytes):
//   [0)          We2T bf16  256*256          = 131072
//   [131072)     Wl1T bf16  256*512          = 262144
//   [393216)     Wl2T bf16  256*256          = 131072
//   [524288)     seg  int   N_REC+1
//   [1048576)    ef   bf16  E*256            = 100663296
//   [101711872)  agg  f32   B*N_REC*512      = 201326592   (total ~303 MB)
// ---------------------------------------------------------------------------
extern "C" void kernel_launch(void* const* d_in, const int* in_sizes, int n_in,
                              void* d_out, int out_size, void* d_ws, size_t ws_size,
                              hipStream_t stream) {
    const float* x         = (const float*)d_in[0];
    const float* edge_attr = (const float*)d_in[1];
    const int*   idx_send  = (const int*)d_in[2];
    const int*   idx_rec   = (const int*)d_in[3];
    const float* We1       = (const float*)d_in[4];
    const float* be1       = (const float*)d_in[5];
    const float* We2       = (const float*)d_in[6];
    const float* be2       = (const float*)d_in[7];
    const float* Wl1       = (const float*)d_in[8];
    const float* bl1       = (const float*)d_in[9];
    const float* Wl2       = (const float*)d_in[10];
    const float* bl2       = (const float*)d_in[11];
    float* out = (float*)d_out;

    char* ws = (char*)d_ws;
    __bf16* We2T = (__bf16*)(ws + 0);
    __bf16* Wl1T = (__bf16*)(ws + 131072);
    __bf16* Wl2T = (__bf16*)(ws + 393216);
    int*    seg  = (int*)(ws + 524288);
    __bf16* ef   = (__bf16*)(ws + 1048576);
    float*  agg  = (float*)(ws + 1048576 + 100663296UL);

    // 0) weight transpose/convert (262144 elements)
    prep_weights_kernel<<<1024, 256, 0, stream>>>(We2, Wl1, Wl2, We2T, Wl1T, Wl2T);
    // 1) segment bounds
    seg_bounds_kernel<<<(N_REC + 1 + 255) / 256, 256, 0, stream>>>(idx_rec, seg);
    // 2) edge MLP: 1536 blocks x 256 threads, 128 edges/block
    edge_mlp_kernel<<<E_TOT / 128, 256, 0, stream>>>(edge_attr, We1, be1, We2T, be2, ef);
    // 3) gather + segment sum: one 64-thread block per receiver
    gather_segsum_kernel<<<N_REC, 64, 0, stream>>>(x, idx_send, ef, seg, agg);
    // 4) node MLP: 1536 blocks x 128 threads, 64 rows/block
    node_mlp_kernel<<<(B_BATCH * N_REC) / 64, 128, 0, stream>>>(agg, Wl1T, bl1, Wl2T, bl2, out);
}
